// AtomAttentionEncoder_47519518163046
// MI455X (gfx1250) — matrix operand order repaired and checked by reading, not compiled, not run on hardware
//
#include <hip/hip_runtime.h>
#include <hip/hip_bf16.h>
#include <math.h>

typedef __attribute__((ext_vector_type(16))) _Float16 v16h;
typedef __attribute__((ext_vector_type(8)))  _Float16 v8h;
typedef __attribute__((ext_vector_type(8)))  float    v8f;
typedef __attribute__((ext_vector_type(4)))  float    v4f;
typedef __attribute__((ext_vector_type(4)))  unsigned int u32x4;
typedef __attribute__((ext_vector_type(8)))  int      i32x8;
typedef __attribute__((ext_vector_type(4)))  int      i32x4;

#define DEV __device__ __forceinline__

// ---- model dims (fixed by the reference) ----
#define BB    2
#define TT    512
#define NN    2048     // atoms per batch
#define AD    128      // ATOM_DIM
#define DP    16
#define TD    384      // TOKEN_DIM
#define DCOND 384
#define PD    128      // PAIR_DIM
#define NQ    32
#define NK    128
#define NHD   4
#define DH    32
#define NB    (NN/NQ)  // 64 query blocks per batch

__constant__ float REFP[4][3] = {
  { 0.8053f, -1.9707f, -1.4140f},
  { 0.7207f, -0.5147f, -1.3373f},
  {-0.0555f, -0.0849f, -0.1298f},
  { 0.5367f,  0.1090f,  0.9654f}};
// ref_feat one-hot column index (5 + elem), elem = {6,5,5,7}
__constant__ int OHCOL[4] = {11, 10, 10, 12};

DEV int clampi(int v, int lo, int hi){ return v < lo ? lo : (v > hi ? hi : v); }
DEV float sigm(float v){ return 1.f/(1.f + __expf(-v)); }

// ------------------- WMMA helpers (gfx1250 layouts) -------------------
DEV v8f wmma_f16(v16h a, v16h b, v8f c){
  return __builtin_amdgcn_wmma_f32_16x16x32_f16(false, a, false, b, (short)0, c, false, false);
}

// A-matrix 16x32 f16 fragment from row-major [16][ldk] tile at K offset k0.
DEV v16h load_a_frag(const _Float16* a, int ldk, int k0){
  const int lane = threadIdx.x & 31;
  const int row  = lane & 15, hi = lane >> 4;
  const _Float16* p = a + row*ldk + k0 + hi*8;
  v8h lo = *(const v8h*)(p);
  v8h hh = *(const v8h*)(p + 16);
  v16h r;
  #pragma unroll
  for (int i = 0; i < 8; i++){ r[i] = lo[i]; r[i+8] = hh[i]; }
  return r;
}

// B-matrix 32x16 f16 fragment from bT[N][ldk] (i.e. W transposed) at K offset k0.
DEV v16h load_b_frag(const _Float16* bT, int ldk, int k0){
  const int lane = threadIdx.x & 31;
  const int col  = lane & 15, hi = lane >> 4;
  const _Float16* p = bT + col*ldk + k0 + hi*8;
  v8h lo = *(const v8h*)(p);
  v8h hh = *(const v8h*)(p + 16);
  v16h r;
  #pragma unroll
  for (int i = 0; i < 8; i++){ r[i] = lo[i]; r[i+8] = hh[i]; }
  return r;
}

// full K=128 16x16 tile
DEV v8f gemm_k128(const _Float16* a, const _Float16* bT, v8f acc){
  #pragma unroll
  for (int k0 = 0; k0 < 128; k0 += 32)
    acc = wmma_f16(load_a_frag(a, 128, k0), load_b_frag(bT, 128, k0), acc);
  return acc;
}

// ------------------- kernel 1: c_tok = LN(cond; g,b) @ W_c -------------------
__global__ __launch_bounds__(256)
void k_cond_proj(const float* __restrict__ cond, const float* __restrict__ lng,
                 const float* __restrict__ lnb, const float* __restrict__ Wc,
                 float* __restrict__ c_tok){
  __shared__ __attribute__((aligned(32))) _Float16 a_ln[16*DCOND];
  __shared__ __attribute__((aligned(32))) _Float16 wTk[128*32];
  const int row0 = blockIdx.x * 16;          // token rows over B*T
  const int tid  = threadIdx.x;
  if (tid < 16){
    const float* r = cond + (size_t)(row0 + tid)*DCOND;
    float mu = 0.f;  for (int j = 0; j < DCOND; j++) mu += r[j];  mu *= (1.f/DCOND);
    float var = 0.f; for (int j = 0; j < DCOND; j++){ float d = r[j]-mu; var += d*d; } var *= (1.f/DCOND);
    float rs = rsqrtf(var + 1e-5f);
    for (int j = 0; j < DCOND; j++)
      a_ln[tid*DCOND + j] = (_Float16)(((r[j]-mu)*rs)*lng[j] + lnb[j]);
  }
  __syncthreads();
  const int wid = tid >> 5, lane = tid & 31, col = lane & 15, hi = lane >> 4;
  v8f acc = {};
  for (int k0 = 0; k0 < DCOND; k0 += 32){
    for (int i = tid; i < 128*32; i += 256){
      int c = i >> 5, kk = i & 31;
      wTk[i] = (_Float16)Wc[(size_t)(k0+kk)*AD + c];
    }
    __syncthreads();
    acc = wmma_f16(load_a_frag(a_ln, DCOND, k0), load_b_frag(wTk + wid*16*32, 32, 0), acc);
    __syncthreads();
  }
  #pragma unroll
  for (int j = 0; j < 8; j++)
    c_tok[(size_t)(row0 + j + 8*hi)*AD + wid*16 + col] = acc[j];
}

// ------------------- kernel 2: atom features + q_l -------------------
__global__ __launch_bounds__(256)
void k_atom_init(const float* __restrict__ coors3d, const float* __restrict__ cmask,
                 const float* __restrict__ Wf, const float* __restrict__ W3d,
                 const float* __restrict__ c_tok,
                 float* __restrict__ c_l, float* __restrict__ x){
  int idx = blockIdx.x*256 + threadIdx.x;          // over B*N*AD
  if (idx >= BB*NN*AD) return;
  int d = idx & (AD-1);
  int g = idx >> 7;                                 // atom global
  int a4 = g & 3, tok = g >> 2;
  float m = cmask[g];
  float c = REFP[a4][0]*Wf[0*AD+d] + REFP[a4][1]*Wf[1*AD+d] + REFP[a4][2]*Wf[2*AD+d]
          + Wf[4*AD+d] + Wf[OHCOL[a4]*AD+d];
  c = c*m + c_tok[(size_t)tok*AD + d];
  float q = c + coors3d[(size_t)g*3+0]*W3d[0*AD+d]
              + coors3d[(size_t)g*3+1]*W3d[1*AD+d]
              + coors3d[(size_t)g*3+2]*W3d[2*AD+d];
  c_l[idx] = c;
  x[idx]   = q;
}

// ------------------- kernel 3: cl = relu(c_l@W_cl), cm = relu(c_l@W_cm) -------------------
__global__ __launch_bounds__(32)
void k_clcm(const float* __restrict__ c_l, const float* __restrict__ Wcl,
            const float* __restrict__ Wcm, float* __restrict__ cl16, float* __restrict__ cm16){
  __shared__ float row[AD];
  int g = blockIdx.x, lane = threadIdx.x;
  for (int j = lane; j < AD; j += 32) row[j] = c_l[(size_t)g*AD + j];
  __syncthreads();
  const float* W = (lane < 16) ? Wcl : Wcm;
  int d = lane & 15;
  float s = 0.f;
  for (int k = 0; k < AD; k++) s += row[k]*W[k*DP + d];
  s = fmaxf(s, 0.f);
  if (lane < 16) cl16[(size_t)g*DP + d] = s; else cm16[(size_t)g*DP + d] = s;
}

// ------------------- kernel 4: build p_lm (zp computed only where gathered) -------------------
__global__ __launch_bounds__(256)
void k_build_plm(const float* __restrict__ pair, const float* __restrict__ cmask,
                 const float* __restrict__ lnzg, const float* __restrict__ lnzb,
                 const float* __restrict__ Wz,
                 const float* __restrict__ Wd, const float* __restrict__ Winv,
                 const float* __restrict__ Wvp,
                 const float* __restrict__ m1, const float* __restrict__ m2,
                 const float* __restrict__ m3,
                 const float* __restrict__ cl16, const float* __restrict__ cm16,
                 float* __restrict__ plm){
  __shared__ float wz[PD*DP];
  __shared__ float zp[256*DP];
  __shared__ float mw0[DP*DP], mw1[DP*DP], mw2[DP*DP];
  __shared__ float wd[3*DP], winv[DP], wv[DP];
  const int b = blockIdx.x >> 6, I = blockIdx.x & 63;
  const int tid = threadIdx.x;
  for (int i = tid; i < PD*DP; i += 256) wz[i] = Wz[i];
  for (int i = tid; i < DP*DP; i += 256){ mw0[i]=m1[i]; mw1[i]=m2[i]; mw2[i]=m3[i]; }
  if (tid < 3*DP) wd[tid] = Wd[tid];
  if (tid < DP){ winv[tid] = Winv[tid]; wv[tid] = Wvp[tid]; }
  __syncthreads();
  const int tq0 = I*8;
  const int ks  = I*NQ - (NK-NQ)/2;
  const int tk0 = clampi(ks, 0, NN-1) >> 2;
  {  // zp for the 8x32 token pairs this block can gather
    int tq = tq0 + (tid >> 5);
    int tk = tk0 + (tid & 31);  if (tk > TT-1) tk = TT-1;
    const float* r = pair + (((size_t)b*TT + tq)*TT + tk)*PD;
    float mu = 0.f;  for (int j = 0; j < PD; j++) mu += r[j];  mu *= (1.f/PD);
    float var = 0.f; for (int j = 0; j < PD; j++){ float dd = r[j]-mu; var += dd*dd; } var *= (1.f/PD);
    float rs = rsqrtf(var + 1e-5f);
    float o[DP];
    for (int d = 0; d < DP; d++) o[d] = 0.f;
    for (int j = 0; j < PD; j++){
      float xn = ((r[j]-mu)*rs)*lnzg[j] + lnzb[j];
      for (int d = 0; d < DP; d++) o[d] += xn * wz[j*DP + d];
    }
    for (int d = 0; d < DP; d++) zp[tid*DP + d] = o[d];
  }
  __syncthreads();
  for (int pidx = tid; pidx < NQ*NK; pidx += 256){
    int q = pidx >> 7, k = pidx & 127;
    int qa = I*NQ + q, g = b*NN + qa;
    int kr = ks + k;
    float valid = (kr >= 0 && kr < NN) ? 1.f : 0.f;
    int ka = clampi(kr, 0, NN-1), kg = b*NN + ka;
    float pm = cmask[g]*cmask[kg]*valid;
    float d0 = (REFP[qa&3][0]-REFP[ka&3][0])*pm;
    float d1 = (REFP[qa&3][1]-REFP[ka&3][1])*pm;
    float d2 = (REFP[qa&3][2]-REFP[ka&3][2])*pm;
    float vv = (((qa>>2)==(ka>>2)) ? 1.f : 0.f)*pm;
    float invd = 1.f/(1.f + d0*d0 + d1*d1 + d2*d2);
    const float* zrow = zp + ((q>>2)*32 + ((ka>>2) - tk0))*DP;
    const float* clr  = cl16 + (size_t)g*DP;
    const float* cmr  = cm16 + (size_t)kg*DP;
    float vec[DP], t0[DP], rr[DP];
    for (int d = 0; d < DP; d++)
      vec[d] = (d0*wd[d] + d1*wd[DP+d] + d2*wd[2*DP+d])*vv
             + invd*winv[d]*vv + wv[d]*vv*vv + zrow[d] + clr[d] + cmr[d];
    for (int k2 = 0; k2 < DP; k2++) rr[k2] = fmaxf(vec[k2], 0.f);
    for (int d = 0; d < DP; d++){ float s = 0.f; for (int k2 = 0; k2 < DP; k2++) s += rr[k2]*mw0[k2*DP+d]; t0[d] = s; }
    for (int k2 = 0; k2 < DP; k2++) rr[k2] = fmaxf(t0[k2], 0.f);
    for (int d = 0; d < DP; d++){ float s = 0.f; for (int k2 = 0; k2 < DP; k2++) s += rr[k2]*mw1[k2*DP+d]; t0[d] = s; }
    for (int k2 = 0; k2 < DP; k2++) rr[k2] = fmaxf(t0[k2], 0.f);
    float* out = plm + ((((size_t)b*NB + I)*NQ + q)*NK + k)*DP;
    for (int d = 0; d < DP; d++){ float s = 0.f; for (int k2 = 0; k2 < DP; k2++) s += rr[k2]*mw2[k2*DP+d]; out[d] = vec[d] + s; }
  }
}

// ------------------- kernel 5: adaLN (two WMMA GEMMs) -------------------
__global__ __launch_bounds__(256)
void k_adaln(const float* __restrict__ X, const float* __restrict__ C,
             const float* __restrict__ lg, const float* __restrict__ lb,
             const float* __restrict__ Wg, const float* __restrict__ bg,
             const float* __restrict__ Wb, const float* __restrict__ cmask,
             float* __restrict__ H){
  __shared__ __attribute__((aligned(32))) _Float16 xn[16*AD];
  __shared__ __attribute__((aligned(32))) _Float16 cn[16*AD];
  __shared__ __attribute__((aligned(32))) _Float16 wT[AD*AD];
  __shared__ float gam[16*AD];
  const int row0 = blockIdx.x*16;
  const int tid = threadIdx.x;
  if (tid < 32){
    int r = tid & 15;
    const float* src = (tid < 16) ? (X + (size_t)(row0+r)*AD) : (C + (size_t)(row0+r)*AD);
    float mu = 0.f;  for (int j = 0; j < AD; j++) mu += src[j];  mu *= (1.f/AD);
    float var = 0.f; for (int j = 0; j < AD; j++){ float d = src[j]-mu; var += d*d; } var *= (1.f/AD);
    float rs = rsqrtf(var + 1e-5f);
    if (tid < 16) for (int j = 0; j < AD; j++) xn[r*AD+j] = (_Float16)((src[j]-mu)*rs);
    else          for (int j = 0; j < AD; j++) cn[r*AD+j] = (_Float16)(((src[j]-mu)*rs)*lg[j] + lb[j]);
  }
  for (int i = tid; i < AD*AD; i += 256) wT[i] = (_Float16)Wg[(size_t)(i&127)*AD + (i>>7)];
  __syncthreads();
  const int wid = tid >> 5, lane = tid & 31, col = lane & 15, hi = lane >> 4;
  v8f acc = {};
  acc = gemm_k128(cn, wT + wid*16*AD, acc);
  #pragma unroll
  for (int j = 0; j < 8; j++){
    int c = wid*16 + col;
    gam[(j + 8*hi)*AD + c] = sigm(acc[j] + bg[c]);
  }
  __syncthreads();
  for (int i = tid; i < AD*AD; i += 256) wT[i] = (_Float16)Wb[(size_t)(i&127)*AD + (i>>7)];
  __syncthreads();
  v8f acc2 = {};
  acc2 = gemm_k128(cn, wT + wid*16*AD, acc2);
  #pragma unroll
  for (int j = 0; j < 8; j++){
    int r = j + 8*hi, c = wid*16 + col;
    float h = gam[r*AD+c]*(float)xn[r*AD+c] + acc2[j];
    H[(size_t)(row0+r)*AD + c] = h*cmask[row0+r];
  }
}

// ------------------- kernel 6: generic K=128 GEMM, 16 rows x 128 cols per block -------------------
// act: 0 none, 1 sigmoid, 2 relu, 3 silu. amul scales A elementwise; omul scales output.
__global__ __launch_bounds__(256)
void k_gemm128(const float* __restrict__ X, const float* __restrict__ amul,
               const float* __restrict__ W, const float* __restrict__ bias,
               const float* __restrict__ omul, int ldw,
               float* __restrict__ out, int act){
  __shared__ __attribute__((aligned(32))) _Float16 a[16*AD];
  __shared__ __attribute__((aligned(32))) _Float16 wT[AD*AD];
  const int row0 = blockIdx.x*16;
  const int col0 = blockIdx.y*128;
  const int tid = threadIdx.x;
  // get the weight panel's cachelines in flight while we convert A
  __builtin_prefetch(&W[(size_t)(tid & 127)*ldw + col0 + ((tid >> 7) << 6)], 0, 0);
  for (int i = tid; i < 16*AD; i += 256){
    size_t idx = (size_t)(row0 + (i>>7))*AD + (i&127);
    float v = X[idx];
    if (amul) v *= amul[idx];
    a[i] = (_Float16)v;
  }
  for (int i = tid; i < AD*128; i += 256)
    wT[i] = (_Float16)W[(size_t)(i&127)*ldw + col0 + (i>>7)];
  __syncthreads();
  const int wid = tid >> 5, lane = tid & 31, col = lane & 15, hi = lane >> 4;
  v8f acc = {};
  acc = gemm_k128(a, wT + wid*16*AD, acc);
  #pragma unroll
  for (int j = 0; j < 8; j++){
    int r = row0 + j + 8*hi;
    int c = col0 + wid*16 + col;
    float v = acc[j] + (bias ? bias[c] : 0.f);
    if (act == 1) v = sigm(v);
    else if (act == 2) v = fmaxf(v, 0.f);
    else if (act == 3) v = v*sigm(v);
    if (omul) v *= omul[(size_t)r*ldw + c];
    out[(size_t)r*ldw + c] = v;
  }
}

// ------------------- kernel 7: per-head LN over DH=32 (full-wave shuffle reduce) -------------------
__global__ __launch_bounds__(32)
void k_headln(float* __restrict__ buf, const float* __restrict__ g, const float* __restrict__ bta){
  int ah = blockIdx.x;               // over B*N*NHD
  int h = ah & 3, atom = ah >> 2;
  int lane = threadIdx.x;
  size_t off = (size_t)atom*AD + h*DH + lane;
  float v = buf[off];
  float mu = v;
  for (int o = 16; o >= 1; o >>= 1) mu += __shfl_xor(mu, o);
  mu *= (1.f/DH);
  float d = v - mu, var = d*d;
  for (int o = 16; o >= 1; o >>= 1) var += __shfl_xor(var, o);
  var *= (1.f/DH);
  buf[off] = d*rsqrtf(var + 1e-5f)*g[lane] + bta[lane];
}

// ------------------- kernel 8: attention logits + bias + mask + softmax -------------------
__global__ __launch_bounds__(256)
void k_attn(const float* __restrict__ qb, const float* __restrict__ kb,
            const float* __restrict__ plm, const float* __restrict__ Wbias,
            const float* __restrict__ cmask, _Float16* __restrict__ attn){
  __shared__ __attribute__((aligned(32))) _Float16 qs[NQ*AD];
  __shared__ __attribute__((aligned(32))) _Float16 ks_[NK*AD];
  __shared__ float qm[NQ], km[NK], wbias[DP*NHD];
  const int b = blockIdx.x >> 6, I = blockIdx.x & 63;
  const int tid = threadIdx.x;
  const int kstart = I*NQ - (NK-NQ)/2;
  for (int i = tid; i < NQ*AD; i += 256)
    qs[i] = (_Float16)qb[((size_t)(b*NN + I*NQ + (i>>7)))*AD + (i&127)];
  for (int i = tid; i < NK*AD; i += 256){
    int ka = clampi(kstart + (i>>7), 0, NN-1);
    ks_[i] = (_Float16)kb[((size_t)(b*NN + ka))*AD + (i&127)];
  }
  if (tid < NQ) qm[tid] = cmask[b*NN + I*NQ + tid];
  if (tid < NK){
    int kr = kstart + tid;
    float valid = (kr >= 0 && kr < NN) ? 1.f : 0.f;
    km[tid] = cmask[b*NN + clampi(kr,0,NN-1)]*valid;
  }
  if (tid < DP*NHD) wbias[tid] = Wbias[tid];
  __syncthreads();
  const int wid = tid >> 5, lane = tid & 31, col = lane & 15, hi = lane >> 4;
  const int h = wid >> 1, r = wid & 1;
  const float scale = 0.17677669529663687f;  // 1/sqrt(DH)
  v16h af = load_a_frag(qs + r*16*AD, AD, h*DH);
  const float* plmb = plm + (((size_t)b*NB + I)*NQ*NK)*DP;
  v8f lg[8];
  #pragma unroll
  for (int ct = 0; ct < 8; ct++){
    v8f z = {};
    v8f acc = wmma_f16(af, load_b_frag(ks_ + ct*16*AD, AD, h*DH), z);
    #pragma unroll
    for (int j = 0; j < 8; j++){
      int q = r*16 + j + 8*hi;
      int k = ct*16 + col;
      const v4f* pr = (const v4f*)(plmb + ((size_t)q*NK + k)*DP);  // 64 B aligned
      float bsum = 0.f;
      #pragma unroll
      for (int t4 = 0; t4 < 4; t4++){
        v4f pv = pr[t4];
        bsum += pv.x*wbias[(t4*4+0)*NHD + h] + pv.y*wbias[(t4*4+1)*NHD + h]
              + pv.z*wbias[(t4*4+2)*NHD + h] + pv.w*wbias[(t4*4+3)*NHD + h];
      }
      float v = acc[j]*scale + bsum;
      if (!(qm[q]*km[k] > 0.f)) v = -1e9f;
      acc[j] = v;
    }
    lg[ct] = acc;
  }
  _Float16* ab = attn + ((((size_t)b*NB + I)*NHD + h)*NQ)*NK;
  #pragma unroll
  for (int j = 0; j < 8; j++){
    float m = -3.4e38f;
    #pragma unroll
    for (int ct = 0; ct < 8; ct++) m = fmaxf(m, lg[ct][j]);
    for (int o = 1; o < 16; o <<= 1) m = fmaxf(m, __shfl_xor(m, o));
    float s = 0.f, e[8];
    #pragma unroll
    for (int ct = 0; ct < 8; ct++){ e[ct] = __expf(lg[ct][j] - m); s += e[ct]; }
    for (int o = 1; o < 16; o <<= 1) s += __shfl_xor(s, o);
    float inv = 1.f/s;
    int q = r*16 + j + 8*hi;
    #pragma unroll
    for (int ct = 0; ct < 8; ct++)
      ab[(size_t)q*NK + ct*16 + col] = (_Float16)(e[ct]*inv);
  }
}

// ------------------- kernel 9: O = attn @ V (TDM-staged A, WMMA) -------------------
__global__ __launch_bounds__(256)
void k_attn_av(const _Float16* __restrict__ attn, const float* __restrict__ vb,
               float* __restrict__ obuf){
  __shared__ __attribute__((aligned(64))) _Float16 as_[NHD*NQ*NK]; // 32 KB
  __shared__ __attribute__((aligned(64))) _Float16 vT[AD*NK];      // 32 KB, vT[d][k]
  const int b = blockIdx.x >> 6, I = blockIdx.x & 63;
  const int tid = threadIdx.x;
  const int kstart = I*NQ - (NK-NQ)/2;
  if (tid == 0){
    // Tensor Data Mover: DMA the contiguous 32 KB attention-probability slab
    // for this (b, I) block into LDS (D# per CDNA5 ISA §8.3/8.4; 1-row tile of
    // 4096 x 8-byte elements). Overlaps with the VALU f32->f16 V staging below.
    unsigned long long ga =
        (unsigned long long)(const void*)(attn + ((size_t)b*NB + I)*NHD*NQ*NK);
    unsigned lds_off = (unsigned)(unsigned long long)(const void*)as_;
    u32x4 g0;
    g0[0] = 1u;                                                  // count=1 user D#
    g0[1] = lds_off;                                             // lds_addr
    g0[2] = (unsigned)(ga & 0xFFFFFFFFull);                      // global_addr[31:0]
    g0[3] = (unsigned)((ga >> 32) & 0x1FFFFFFull) | (2u << 30);  // addr[56:32], type=2
    i32x8 g1 = {};
    g1[0] = 0x00030000;           // data_size = 8 bytes
    g1[1] = (int)(4096u << 16);   // tensor_dim0 = 4096 (low 16 bits at [63:48])
    g1[2] = (int)(1u << 16);      // tensor_dim0 hi = 0 | tensor_dim1 = 1
    g1[3] = (int)(4096u << 16);   // tile_dim0 = 4096
    g1[4] = 1;                    // tile_dim1 = 1
    g1[5] = 4096;                 // tensor_dim0_stride (elements)
    i32x4 z4 = {};
    i32x8 z8 = {};
    __builtin_amdgcn_tensor_load_to_lds(g0, g1, z4, z4, z8, 0);
  }
  for (int i = tid; i < AD*NK; i += 256){
    int ka = clampi(kstart + (i&127), 0, NN-1);
    vT[i] = (_Float16)vb[((size_t)(b*NN + ka))*AD + (i>>7)];
  }
  if (tid == 0) __builtin_amdgcn_s_wait_tensorcnt(0);
  __syncthreads();
  const int wid = tid >> 5, lane = tid & 31, col = lane & 15, hi = lane >> 4;
  const int h = wid >> 1, r = wid & 1;
  const _Float16* ab = as_ + ((size_t)h*NQ + r*16)*NK;
  #pragma unroll
  for (int nt = 0; nt < 2; nt++){
    v8f acc = {};
    #pragma unroll
    for (int k0 = 0; k0 < NK; k0 += 32)
      acc = wmma_f16(load_a_frag(ab, NK, k0), load_b_frag(vT + (h*DH + nt*16)*NK, NK, k0), acc);
    #pragma unroll
    for (int j = 0; j < 8; j++){
      int atom = b*NN + I*NQ + r*16 + j + 8*hi;
      obuf[(size_t)atom*AD + h*DH + nt*16 + col] = acc[j];
    }
  }
}

// ------------------- kernel 10: K=512 GEMM (transition down-proj) -------------------
__global__ __launch_bounds__(256)
void k_gemm_k512(const float* __restrict__ X, const float* __restrict__ W,
                 float* __restrict__ out){
  __shared__ __attribute__((aligned(32))) _Float16 a[16*128];
  __shared__ __attribute__((aligned(32))) _Float16 wT[128*128];
  const int row0 = blockIdx.x*16;
  const int tid = threadIdx.x;
  const int wid = tid >> 5, lane = tid & 31, col = lane & 15, hi = lane >> 4;
  v8f acc = {};
  for (int kc = 0; kc < 4; kc++){
    __syncthreads();
    for (int i = tid; i < 16*128; i += 256)
      a[i] = (_Float16)X[(size_t)(row0 + (i>>7))*512 + kc*128 + (i&127)];
    for (int i = tid; i < 128*128; i += 256)
      wT[i] = (_Float16)W[(size_t)(kc*128 + (i&127))*AD + (i>>7)];
    __syncthreads();
    acc = gemm_k128(a, wT + wid*16*AD, acc);
  }
  #pragma unroll
  for (int j = 0; j < 8; j++)
    out[(size_t)(row0 + j + 8*hi)*AD + wid*16 + col] = acc[j];
}

// ------------------- kernel 11: residual x = (s*o*m + x)*m -------------------
__global__ __launch_bounds__(256)
void k_residual(const float* __restrict__ s, const float* __restrict__ o,
                const float* __restrict__ cmask, float* __restrict__ x){
  int idx = blockIdx.x*256 + threadIdx.x;
  if (idx >= BB*NN*AD) return;
  float m = cmask[idx >> 7];
  x[idx] = (s[idx]*o[idx]*m + x[idx])*m;
}

// ------------------- kernel 12: token head (WMMA + fused masked mean over 4 atoms) -------------------
__global__ __launch_bounds__(256)
void k_token(const float* __restrict__ X, const float* __restrict__ Wq,
             const float* __restrict__ cmask, const float* __restrict__ mask,
             float* __restrict__ out){
  __shared__ __attribute__((aligned(32))) _Float16 a[16*AD];
  __shared__ __attribute__((aligned(32))) _Float16 wT[128*AD];
  __shared__ float rcm[16];
  const int row0 = blockIdx.x*16;          // atom rows (16 atoms = 4 tokens)
  const int col0 = blockIdx.y*128;
  const int tid = threadIdx.x;
  for (int i = tid; i < 16*AD; i += 256)
    a[i] = (_Float16)X[(size_t)(row0 + (i>>7))*AD + (i&127)];
  for (int i = tid; i < 128*AD; i += 256)
    wT[i] = (_Float16)Wq[(size_t)(i&127)*TD + col0 + (i>>7)];
  if (tid < 16) rcm[tid] = cmask[row0 + tid];
  __syncthreads();
  const int wid = tid >> 5, lane = tid & 31, col = lane & 15, hi = lane >> 4;
  v8f acc = {};
  acc = gemm_k128(a, wT + wid*16*AD, acc);
  int c = col0 + wid*16 + col;
  float s0 = 0.f, d0 = 0.f, s1 = 0.f, d1 = 0.f;
  #pragma unroll
  for (int j = 0; j < 4; j++){
    float m0 = rcm[j + 8*hi];
    s0 += fmaxf(acc[j], 0.f)*m0;  d0 += m0;
    float m1v = rcm[4 + j + 8*hi];
    s1 += fmaxf(acc[4+j], 0.f)*m1v;  d1 += m1v;
  }
  int t0 = (row0 >> 2) + 2*hi;   // token index over B*T
  out[(size_t)t0*TD + c]     = s0 / fmaxf(d0, 1.f) * mask[t0];
  out[(size_t)(t0+1)*TD + c] = s1 / fmaxf(d1, 1.f) * mask[t0+1];
}

// ------------------- host orchestration -------------------
extern "C" void kernel_launch(void* const* d_in, const int* in_sizes, int n_in,
                              void* d_out, int out_size, void* d_ws, size_t ws_size,
                              hipStream_t stream){
  (void)in_sizes; (void)n_in; (void)out_size; (void)ws_size;
  const float* coors3d = (const float*)d_in[0];
  const float* pair    = (const float*)d_in[1];
  const float* cond    = (const float*)d_in[2];
  const float* cmask   = (const float*)d_in[3];
  const float* mask    = (const float*)d_in[4];
  const float* W_f = (const float*)d_in[5];
  const float* W_d = (const float*)d_in[6];
  const float* W_invd = (const float*)d_in[7];
  const float* W_v = (const float*)d_in[8];
  const float* ln_c_g = (const float*)d_in[9];
  const float* ln_c_b = (const float*)d_in[10];
  const float* W_c = (const float*)d_in[11];
  const float* ln_z_g = (const float*)d_in[12];
  const float* ln_z_b = (const float*)d_in[13];
  const float* W_z = (const float*)d_in[14];
  const float* W_3d = (const float*)d_in[15];
  const float* W_cl = (const float*)d_in[16];
  const float* W_cm = (const float*)d_in[17];
  const float* mlp_W1 = (const float*)d_in[18];
  const float* mlp_W2 = (const float*)d_in[19];
  const float* mlp_W3 = (const float*)d_in[20];
  const float* W_q = (const float*)d_in[21];
  const float* a1_lnc_g = (const float*)d_in[22];
  const float* a1_lnc_b = (const float*)d_in[23];
  const float* a1_Wg = (const float*)d_in[24];
  const float* a1_bg = (const float*)d_in[25];
  const float* a1_Wb = (const float*)d_in[26];
  const float* att_Wq = (const float*)d_in[27];
  const float* att_bq = (const float*)d_in[28];
  const float* att_Wk = (const float*)d_in[29];
  const float* att_bk = (const float*)d_in[30];
  const float* att_Wv = (const float*)d_in[31];
  const float* att_bv = (const float*)d_in[32];
  const float* att_qln_g = (const float*)d_in[33];
  const float* att_qln_b = (const float*)d_in[34];
  const float* att_kln_g = (const float*)d_in[35];
  const float* att_kln_b = (const float*)d_in[36];
  const float* att_Wbias = (const float*)d_in[37];
  const float* att_Wgate = (const float*)d_in[38];
  const float* att_bgate = (const float*)d_in[39];
  const float* att_Wo = (const float*)d_in[40];
  const float* att_bo = (const float*)d_in[41];
  const float* s1_Ws = (const float*)d_in[42];
  const float* s1_bs = (const float*)d_in[43];
  const float* a2_lnc_g = (const float*)d_in[44];
  const float* a2_lnc_b = (const float*)d_in[45];
  const float* a2_Wg = (const float*)d_in[46];
  const float* a2_bg = (const float*)d_in[47];
  const float* a2_Wb = (const float*)d_in[48];
  const float* tr_Wa = (const float*)d_in[49];
  const float* tr_Wb = (const float*)d_in[50];
  const float* tr_Wo = (const float*)d_in[51];
  const float* s2_Ws = (const float*)d_in[52];
  const float* s2_bs = (const float*)d_in[53];

  float* ws = (float*)d_ws;
  size_t off = 0;
  auto alloc = [&](size_t n)->float*{ float* p = ws + off; off += (n + 63) & ~(size_t)63; return p; };
  float* c_tok = alloc((size_t)BB*TT*AD);
  float* c_l   = alloc((size_t)BB*NN*AD);
  float* x     = alloc((size_t)BB*NN*AD);
  float* h     = alloc((size_t)BB*NN*AD);
  float* qb    = alloc((size_t)BB*NN*AD);
  float* kb    = alloc((size_t)BB*NN*AD);
  float* vb    = alloc((size_t)BB*NN*AD);
  float* gb    = alloc((size_t)BB*NN*AD);
  float* obuf  = alloc((size_t)BB*NN*AD);
  float* o2    = alloc((size_t)BB*NN*AD);
  float* sbuf  = alloc((size_t)BB*NN*AD);
  float* cl16  = alloc((size_t)BB*NN*DP);
  float* cm16  = alloc((size_t)BB*NN*DP);
  float* plm   = alloc((size_t)BB*NB*NQ*NK*DP);
  _Float16* attn = (_Float16*)alloc(((size_t)BB*NB*NHD*NQ*NK)/2);
  float* ta    = alloc((size_t)BB*NN*512);
  float* tb    = alloc((size_t)BB*NN*512);

  const int RT = BB*NN/16;       // 256 row tiles
  k_cond_proj<<<BB*TT/16, 256, 0, stream>>>(cond, ln_c_g, ln_c_b, W_c, c_tok);
  k_atom_init<<<(BB*NN*AD)/256, 256, 0, stream>>>(coors3d, cmask, W_f, W_3d, c_tok, c_l, x);
  k_clcm<<<BB*NN, 32, 0, stream>>>(c_l, W_cl, W_cm, cl16, cm16);
  k_build_plm<<<BB*NB, 256, 0, stream>>>(pair, cmask, ln_z_g, ln_z_b, W_z,
                                         W_d, W_invd, W_v, mlp_W1, mlp_W2, mlp_W3,
                                         cl16, cm16, plm);
  for (int i = 0; i < 3; i++){
    const size_t sW = (size_t)AD*AD, sB = AD, sH = DH, sBi = (size_t)DP*NHD, sT = (size_t)AD*512;
    k_adaln<<<RT, 256, 0, stream>>>(x, c_l, a1_lnc_g + i*sB, a1_lnc_b + i*sB,
                                    a1_Wg + i*sW, a1_bg + i*sB, a1_Wb + i*sW, cmask, h);
    dim3 g1(RT, 1);
    k_gemm128<<<g1, 256, 0, stream>>>(h, nullptr, att_Wq + i*sW, att_bq + i*sB, nullptr, AD, qb, 0);
    k_gemm128<<<g1, 256, 0, stream>>>(h, nullptr, att_Wk + i*sW, att_bk + i*sB, nullptr, AD, kb, 0);
    k_gemm128<<<g1, 256, 0, stream>>>(h, nullptr, att_Wv + i*sW, att_bv + i*sB, nullptr, AD, vb, 0);
    k_gemm128<<<g1, 256, 0, stream>>>(h, nullptr, att_Wgate + i*sW, att_bgate + i*sB, nullptr, AD, gb, 1);
    k_headln<<<BB*NN*NHD, 32, 0, stream>>>(qb, att_qln_g + i*sH, att_qln_b + i*sH);
    k_headln<<<BB*NN*NHD, 32, 0, stream>>>(kb, att_kln_g + i*sH, att_kln_b + i*sH);
    k_attn<<<BB*NB, 256, 0, stream>>>(qb, kb, plm, att_Wbias + i*sBi, cmask, attn);
    k_attn_av<<<BB*NB, 256, 0, stream>>>(attn, vb, obuf);
    k_gemm128<<<g1, 256, 0, stream>>>(obuf, gb, att_Wo + i*sW, att_bo + i*sB, nullptr, AD, o2, 0);
    k_gemm128<<<g1, 256, 0, stream>>>(c_l, nullptr, s1_Ws + i*sW, s1_bs + i*sB, nullptr, AD, sbuf, 1);
    k_residual<<<(BB*NN*AD)/256, 256, 0, stream>>>(sbuf, o2, cmask, x);
    k_adaln<<<RT, 256, 0, stream>>>(x, c_l, a2_lnc_g + i*sB, a2_lnc_b + i*sB,
                                    a2_Wg + i*sW, a2_bg + i*sB, a2_Wb + i*sW, cmask, h);
    dim3 g4(RT, 4);
    k_gemm128<<<g4, 256, 0, stream>>>(h, nullptr, tr_Wa + i*sT, nullptr, nullptr, 512, ta, 3);
    k_gemm128<<<g4, 256, 0, stream>>>(h, nullptr, tr_Wb + i*sT, nullptr, ta, 512, tb, 0);
    k_gemm_k512<<<RT, 256, 0, stream>>>(tb, tr_Wo + i*sT, o2);
    k_gemm128<<<g1, 256, 0, stream>>>(c_l, nullptr, s2_Ws + i*sW, s2_bs + i*sB, nullptr, AD, sbuf, 1);
    k_residual<<<(BB*NN*AD)/256, 256, 0, stream>>>(sbuf, o2, cmask, x);
  }
  dim3 gt(RT, 3);
  k_token<<<gt, 256, 0, stream>>>(x, W_q, cmask, mask, (float*)d_out);
}
